// EncoderWav_70858370450056
// MI455X (gfx1250) — compile-verified, matrix operand order checked by reading, not run on hardware
//
#include <hip/hip_runtime.h>
#include <hip/hip_bf16.h>

// ---------------------------------------------------------------------------
// CDNA5 (gfx1250, wave32) wavelet residual encoder.
// Convolutions: implicit GEMM on v_wmma_f32_16x16x32_bf16 with NHWC
// channel-padded activations (Cpad % 32 == 0) and [Cout][Kpad] bf16 weights,
// so A/B staging is contiguous b128 traffic and fragments are 2x b128 LDS
// loads each. All hot-path index math is shift/mask (H,W,HW,Cout are pow2).
// ---------------------------------------------------------------------------

typedef __attribute__((ext_vector_type(16))) __bf16 bf16x16;
typedef __attribute__((ext_vector_type(8)))  __bf16 bf16x8;
typedef __attribute__((ext_vector_type(8)))  float floatx8;

#define BN_EPS 1e-5f

static __device__ inline __bf16 f2bf(float f) {
  unsigned u = __builtin_bit_cast(unsigned, f);
  unsigned r = (u + 0x7FFFu + ((u >> 16) & 1u)) >> 16;   // RNE
  unsigned short s = (unsigned short)r;
  __bf16 b;
  __builtin_memcpy(&b, &s, 2);
  return b;
}

static __device__ inline bf16x8 bf8_zero() {
  union { bf16x8 v; unsigned long long u[2]; } z;
  z.u[0] = 0ull; z.u[1] = 0ull;
  return z.v;
}

// ---------------------------------------------------------------------------
// Build block input in NHWC-padded bf16: channels [0,Cpool)=maxpool2(Eprev),
// [Cpool,Cpool+12)=Haar subbands of LLprev (order c*4+{LL,LH,HL,HH}),
// [Cin,Cpad)=zero. One wave per output pixel; lanes stride channels.
// Also emits LL_k fp32 for the next level.
// ---------------------------------------------------------------------------
__global__ __launch_bounds__(256) void build_input_kernel(
    const float* __restrict__ Eprev, const float* __restrict__ LLprev,
    __bf16* __restrict__ Xin, float* __restrict__ LLout,
    int Cpool, int Cpad, int sW)
{
  const int Cin = Cpool + 12;
  const int W = 1 << sW, Wm1 = W - 1;
  const int pix = blockIdx.x * 8 + (threadIdx.x >> 5);
  const int lane = threadIdx.x & 31;
  const int w = pix & Wm1;
  const int h = (pix >> sW) & Wm1;
  const int n = pix >> (2 * sW);
  const int H2W2row = 2 * W;           // row stride of the 2H x 2W source
  __bf16* xrow = Xin + (size_t)pix * Cpad;

  for (int c = lane; c < Cpad; c += 32) {
    float val = 0.f;
    if (c < Cpool) {
      const float* p = Eprev + (((size_t)(n * Cpool + c) * 2 * W + 2 * h) * 2 * W) + 2 * w;
      val = fmaxf(fmaxf(p[0], p[1]), fmaxf(p[H2W2row], p[H2W2row + 1]));
    } else if (c < Cin) {
      int cw = c - Cpool; int cs = cw >> 2; int s = cw & 3;
      const float* p = LLprev + (((size_t)(n * 3 + cs) * 2 * W + 2 * h) * 2 * W) + 2 * w;
      float a = p[0], b = p[1], cc = p[H2W2row], d = p[H2W2row + 1];
      float LL = (a + b + cc + d) * 0.5f;
      float LH = (a + b - cc - d) * 0.5f;
      float HL = (a - b + cc - d) * 0.5f;
      float HH = (a - b - cc + d) * 0.5f;
      val = (s == 0) ? LL : (s == 1) ? LH : (s == 2) ? HL : HH;
      if (s == 0 && LLout)
        LLout[((((size_t)(n * 3 + cs) << sW) + h) << sW) + w] = LL;
    }
    xrow[c] = f2bf(val);
  }
}

// ---------------------------------------------------------------------------
// Weight transform: fp32 OIHW -> bf16 [co][Kpad], K = tap*Cpad + ci,
// zero for ci >= Cin. grid = (ceil(Cpad/256), taps, Cout).
// ---------------------------------------------------------------------------
__global__ __launch_bounds__(256) void wt_transform_kernel(
    const float* __restrict__ Wsrc, __bf16* __restrict__ Wt,
    int Cin, int Cpad, int KW, int Kpad)
{
  int ci  = blockIdx.x * 256 + threadIdx.x;
  if (ci >= Cpad) return;
  int tap = blockIdx.y;
  int co  = blockIdx.z;
  int kh = tap / KW, kw = tap % KW;      // uniform scalar
  float f = 0.f;
  if (ci < Cin)
    f = Wsrc[(((size_t)co * Cin + ci) * KW + kh) * KW + kw];
  Wt[(size_t)co * Kpad + (size_t)tap * Cpad + ci] = f2bf(f);
}

// ---------------------------------------------------------------------------
// Implicit-GEMM conv via WMMA bf16.
//   X: NHWC-padded bf16 [B,H,W,Cpad]; Wt: bf16 [Cout][Kpad]; Y: NCHW fp32.
// Workgroup 256 thr = 8 waves (4m x 2n), block tile 128x64, wave tile 32x32
// (4 accumulators, 4 WMMA per 32-K chunk). Every 32-K chunk lies in one
// filter tap (Cpad % 32 == 0), so hi/wi are uniform per chunk.
// ---------------------------------------------------------------------------
#define TBM 128
#define TBN 64
#define TBK 32

__global__ __launch_bounds__(256) void conv_gemm_bf16(
    const __bf16* __restrict__ X, const __bf16* __restrict__ Wt,
    const float* __restrict__ bias, float* __restrict__ Y,
    int Cpad, int sW, int Cout, int Kpad, int KW, int pad)
{
  __shared__ __bf16 lA[TBM * TBK];   // [row][k]   8 KB
  __shared__ __bf16 lB[TBN * TBK];   // [n][k]     4 KB

  const int W = 1 << sW, Wm1 = W - 1;
  const int tid  = threadIdx.x;
  const int wave = tid >> 5, lane = tid & 31;
  const int wm = wave >> 1, wn = wave & 1;        // 4 x 2 waves
  const int m0 = blockIdx.x * TBM;
  const int n0 = blockIdx.y * TBN;
  const int nloc = lane & 15, hh = lane >> 4;     // lane half

  floatx8 c00 = {}, c01 = {}, c10 = {}, c11 = {};

  int kh = -pad, kw = -pad, cbase = 0;            // kh/kw pre-offset by pad
  for (int k0 = 0; k0 < Kpad; k0 += TBK) {
    // ---- stage A: 128 rows x 32 ch, 16B chunks (2 per thread) ----
#pragma unroll
    for (int l = tid; l < TBM * 4; l += 256) {
      int row = l >> 2, q = l & 3;
      int m = m0 + row;
      int wo = m & Wm1, ho = (m >> sW) & Wm1, n = m >> (2 * sW);
      int hi = ho + kh, wi = wo + kw;
      bf16x8 v = bf8_zero();
      if ((unsigned)hi < (unsigned)W && (unsigned)wi < (unsigned)W)
        v = *(const bf16x8*)(X + ((size_t)(((n << sW) + hi) << sW) + wi) * Cpad
                               + cbase + q * 8);
      *(bf16x8*)(lA + row * TBK + q * 8) = v;
    }
    // ---- stage B: 64 n-rows x 32 k (1 chunk per thread) ----
    {
      int row = tid >> 2, q = tid & 3;
      *(bf16x8*)(lB + row * TBK + q * 8) =
          *(const bf16x8*)(Wt + (size_t)(n0 + row) * Kpad + k0 + q * 8);
    }
    __syncthreads();

    // ---- fragments: ISA layout == two contiguous b128 LDS loads each ----
    union Frag { bf16x16 v; bf16x8 h[2]; };
    Frag a0, a1, b0, b1;
    const __bf16* pa0 = lA + (wm * 32 + nloc) * TBK + hh * 8;
    const __bf16* pa1 = pa0 + 16 * TBK;
    const __bf16* pb0 = lB + (wn * 32 + nloc) * TBK + hh * 8;
    const __bf16* pb1 = pb0 + 16 * TBK;
    a0.h[0] = *(const bf16x8*)(pa0);      a0.h[1] = *(const bf16x8*)(pa0 + 16);
    a1.h[0] = *(const bf16x8*)(pa1);      a1.h[1] = *(const bf16x8*)(pa1 + 16);
    b0.h[0] = *(const bf16x8*)(pb0);      b0.h[1] = *(const bf16x8*)(pb0 + 16);
    b1.h[0] = *(const bf16x8*)(pb1);      b1.h[1] = *(const bf16x8*)(pb1 + 16);

    c00 = __builtin_amdgcn_wmma_f32_16x16x32_bf16(false, a0.v, false, b0.v, (short)0, c00, false, false);
    c01 = __builtin_amdgcn_wmma_f32_16x16x32_bf16(false, a0.v, false, b1.v, (short)0, c01, false, false);
    c10 = __builtin_amdgcn_wmma_f32_16x16x32_bf16(false, a1.v, false, b0.v, (short)0, c10, false, false);
    c11 = __builtin_amdgcn_wmma_f32_16x16x32_bf16(false, a1.v, false, b1.v, (short)0, c11, false, false);
    __syncthreads();

    // uniform tap walk (SALU): next 32 channels, wrap to next filter tap
    cbase += TBK;
    if (cbase == Cpad) {
      cbase = 0;
      if (++kw == KW - pad) { kw = -pad; ++kh; }
    }
  }

  // ---- store NCHW fp32 (+bias). D layout: n=lane%16, m=vgpr+8*(lane/16) ----
  const int co0 = n0 + wn * 32 + nloc;
  const int co1 = co0 + 16;
  const float bs0 = bias[co0], bs1 = bias[co1];
#pragma unroll
  for (int r = 0; r < 8; ++r) {
    int mA = m0 + wm * 32 + r + 8 * hh;
    int mB = mA + 16;
    int woA = mA & Wm1, hoA = (mA >> sW) & Wm1, nA = mA >> (2 * sW);
    int woB = mB & Wm1, hoB = (mB >> sW) & Wm1, nB = mB >> (2 * sW);
    size_t iA0 = ((size_t)(nA * Cout + co0) << (2 * sW)) + (hoA << sW) + woA;
    size_t iA1 = ((size_t)(nA * Cout + co1) << (2 * sW)) + (hoA << sW) + woA;
    size_t iB0 = ((size_t)(nB * Cout + co0) << (2 * sW)) + (hoB << sW) + woB;
    size_t iB1 = ((size_t)(nB * Cout + co1) << (2 * sW)) + (hoB << sW) + woB;
    Y[iA0] = c00[r] + bs0;
    Y[iA1] = c01[r] + bs1;
    Y[iB0] = c10[r] + bs0;
    Y[iB1] = c11[r] + bs1;
  }
}

// ---------------------------------------------------------------------------
// BatchNorm statistics: per-channel sum/sumsq, block reduce + atomics.
// ---------------------------------------------------------------------------
__global__ __launch_bounds__(256) void bn_stats_kernel(
    const float* __restrict__ Y, float* __restrict__ stats,
    int C, int B, int sHW)
{
  const int c = blockIdx.x;
  const int HW = 1 << sHW;
  const long P = (long)B << sHW;
  const long chunk = (P + gridDim.y - 1) / gridDim.y;
  long i0 = (long)blockIdx.y * chunk;
  long i1 = i0 + chunk; if (i1 > P) i1 = P;
  float sum = 0.f, sq = 0.f;
  for (long i = i0 + threadIdx.x; i < i1; i += blockDim.x) {
    long n = i >> sHW, j = i & (HW - 1);
    float v = Y[(((long)n * C + c) << sHW) + j];
    sum += v; sq += v * v;
  }
  __shared__ float sA[256], sB[256];
  sA[threadIdx.x] = sum; sB[threadIdx.x] = sq;
  __syncthreads();
  for (int s = 128; s > 0; s >>= 1) {
    if (threadIdx.x < s) { sA[threadIdx.x] += sA[threadIdx.x + s];
                           sB[threadIdx.x] += sB[threadIdx.x + s]; }
    __syncthreads();
  }
  if (threadIdx.x == 0) {
    atomicAdd(&stats[2 * c],     sA[0]);
    atomicAdd(&stats[2 * c + 1], sB[0]);
  }
}

__global__ __launch_bounds__(256) void zero_kernel(float* __restrict__ p, int n)
{
  int i = blockIdx.x * blockDim.x + threadIdx.x;
  if (i < n) p[i] = 0.f;
}

// a = bf16(relu(bn(y))) ; y is NCHW fp32, a is NHWC bf16 (C pow2)
__global__ __launch_bounds__(256) void bn_relu_bf16_kernel(
    const float* __restrict__ Y, const float* __restrict__ stats,
    const float* __restrict__ gamma, const float* __restrict__ beta,
    __bf16* __restrict__ A, int sC, int sHW, long total, float invP)
{
  long i = (long)blockIdx.x * blockDim.x + threadIdx.x;
  if (i >= total) return;
  const int C = 1 << sC, HW = 1 << sHW;
  int j = (int)(i & (HW - 1));
  int c = (int)((i >> sHW) & (C - 1));
  int n = (int)(i >> (sHW + sC));
  float mean = stats[2 * c] * invP;
  float var  = stats[2 * c + 1] * invP - mean * mean;
  float v = gamma[c] * (Y[i] - mean) * rsqrtf(var + BN_EPS) + beta[c];
  A[((((size_t)n << sHW) + j) << sC) + c] = f2bf(fmaxf(v, 0.f));
}

// e = relu(bn2(y2) + bns(ys)) -> NCHW fp32 output
__global__ __launch_bounds__(256) void bn_final_kernel(
    const float* __restrict__ Y2, const float* __restrict__ Ys,
    const float* __restrict__ st2, const float* __restrict__ stS,
    const float* __restrict__ g2,  const float* __restrict__ be2,
    const float* __restrict__ gs,  const float* __restrict__ bes,
    float* __restrict__ out, int sC, int sHW, long total, float invP)
{
  long i = (long)blockIdx.x * blockDim.x + threadIdx.x;
  if (i >= total) return;
  const int C = 1 << sC;
  int c = (int)((i >> sHW) & (C - 1));
  float m2 = st2[2 * c] * invP;
  float v2 = st2[2 * c + 1] * invP - m2 * m2;
  float ms = stS[2 * c] * invP;
  float vs = stS[2 * c + 1] * invP - ms * ms;
  float a = g2[c] * (Y2[i] - m2) * rsqrtf(v2 + BN_EPS) + be2[c];
  float b = gs[c] * (Ys[i] - ms) * rsqrtf(vs + BN_EPS) + bes[c];
  out[i] = fmaxf(a + b, 0.f);
}

// ---------------------------------------------------------------------------
// Host orchestration
// ---------------------------------------------------------------------------
extern "C" void kernel_launch(void* const* d_in, const int* in_sizes, int n_in,
                              void* d_out, int out_size, void* d_ws, size_t ws_size,
                              hipStream_t stream) {
  (void)in_sizes; (void)n_in; (void)out_size; (void)ws_size;

  const int Bsz = 8;
  const float* x_img = (const float*)d_in[0];
  auto PRM = [&](int blk, int j) -> const float* {
    return (const float*)d_in[1 + blk * 12 + j];   // w1,b1,g1,be1,w2,b2,g2,be2,ws,bs,gs,bes
  };

  static const int Cpool[5] = {3, 64, 128, 256, 512};
  static const int CoutA[5] = {64, 128, 256, 512, 1024};
  static const int sCA[5]   = {6, 7, 8, 9, 10};
  static const int sWA[5]   = {8, 7, 6, 5, 4};

  // ---- workspace carve-out ----
  char* ws = (char*)d_ws;
  size_t off = 0;
  auto take = [&](size_t bytes) -> char* {
    off = (off + 255) & ~(size_t)255;
    char* p = ws + off;
    off += bytes;
    return p;
  };
  float* LL[4];
  LL[0] = (float*)take((size_t)Bsz * 3 * 256 * 256 * 4);
  LL[1] = (float*)take((size_t)Bsz * 3 * 128 * 128 * 4);
  LL[2] = (float*)take((size_t)Bsz * 3 * 64 * 64 * 4);
  LL[3] = (float*)take((size_t)Bsz * 3 * 32 * 32 * 4);
  __bf16* xin = (__bf16*)take((size_t)16777216 * 2);  // max B*HW*Cpad (blk1: 524288*32)
  __bf16* a1  = (__bf16*)take((size_t)33554432 * 2);  // max B*HW*Cout (blk1)
  __bf16* Wt  = (__bf16*)take((size_t)9437184 * 2);   // max Cout*Kpad (blk5 conv2)
  float*  yA  = (float*)take((size_t)33554432 * 4);   // y1 then reused as y2
  float*  ysB = (float*)take((size_t)33554432 * 4);   // shortcut conv out
  float*  stats = (float*)take((size_t)3 * 2048 * 4); // st1 | st2 | stS
  float* st1 = stats, *st2 = stats + 2048, *stS = stats + 4096;

  // ---- output layout (fp32, concatenated: x_img, e1..e5) ----
  const size_t xelems = (size_t)Bsz * 3 * 512 * 512;
  size_t e_off[5]; size_t cur = xelems;
  for (int k = 0; k < 5; ++k) {
    e_off[k] = cur;
    cur += (size_t)Bsz * CoutA[k] << (2 * sWA[k]);
  }
  (void)hipMemcpyAsync(d_out, d_in[0], xelems * sizeof(float),
                       hipMemcpyDeviceToDevice, stream);

  for (int k = 0; k < 5; ++k) {
    const int sW = sWA[k], sHW = 2 * sW, sC = sCA[k];
    const int Cp = Cpool[k], Ci = Cp + 12, Co = CoutA[k];
    const int Cpad = (Ci + 31) & ~31;
    const int Kpad1 = 9 * Cpad;        // conv1 K
    const int Kpad2 = 9 * Co;          // conv2 K (Co already mult of 32)
    const int KpadS = Cpad;            // shortcut K
    const long totalOut = (long)Bsz * Co << sHW;
    const float invP = 1.f / (float)((long)Bsz << sHW);
    const int M = Bsz << sHW;

    const float* Eprev  = (k == 0) ? x_img : ((float*)d_out + e_off[k - 1]);
    const float* LLprev = (k == 0) ? x_img : LL[k - 1];
    float* LLout = (k < 4) ? LL[k] : nullptr;

    // 1) concat(maxpool(prev), haar) -> NHWC-padded bf16 input (+ LL_k)
    build_input_kernel<<<dim3((unsigned)((Bsz << sHW) / 8)), 256, 0, stream>>>(
        Eprev, LLprev, xin, LLout, Cp, Cpad, sW);

    zero_kernel<<<dim3(24), 256, 0, stream>>>(stats, 3 * 2048);

    // 2) conv1 (3x3, pad1)
    wt_transform_kernel<<<dim3((Cpad + 255) / 256, 9, Co), 256, 0, stream>>>(
        PRM(k, 0), Wt, Ci, Cpad, 3, Kpad1);
    conv_gemm_bf16<<<dim3(M / TBM, Co / TBN), 256, 0, stream>>>(
        xin, Wt, PRM(k, 1), yA, Cpad, sW, Co, Kpad1, 3, 1);
    bn_stats_kernel<<<dim3(Co, 16), 256, 0, stream>>>(yA, st1, Co, Bsz, sHW);
    bn_relu_bf16_kernel<<<dim3((unsigned)((totalOut + 255) / 256)), 256, 0, stream>>>(
        yA, st1, PRM(k, 2), PRM(k, 3), a1, sC, sHW, totalOut, invP);

    // 3) conv2 (3x3, pad1): a1 -> yA (reuse; y1 dead after apply)
    wt_transform_kernel<<<dim3((Co + 255) / 256, 9, Co), 256, 0, stream>>>(
        PRM(k, 4), Wt, Co, Co, 3, Kpad2);
    conv_gemm_bf16<<<dim3(M / TBM, Co / TBN), 256, 0, stream>>>(
        a1, Wt, PRM(k, 5), yA, Co, sW, Co, Kpad2, 3, 1);
    bn_stats_kernel<<<dim3(Co, 16), 256, 0, stream>>>(yA, st2, Co, Bsz, sHW);

    // 4) shortcut conv (1x1, pad0): xin -> ysB
    wt_transform_kernel<<<dim3((Cpad + 255) / 256, 1, Co), 256, 0, stream>>>(
        PRM(k, 8), Wt, Ci, Cpad, 1, KpadS);
    conv_gemm_bf16<<<dim3(M / TBM, Co / TBN), 256, 0, stream>>>(
        xin, Wt, PRM(k, 9), ysB, Cpad, sW, Co, KpadS, 1, 0);
    bn_stats_kernel<<<dim3(Co, 16), 256, 0, stream>>>(ysB, stS, Co, Bsz, sHW);

    // 5) e_k = relu(bn2(y2) + bns(ys)) -> fp32 output
    bn_final_kernel<<<dim3((unsigned)((totalOut + 255) / 256)), 256, 0, stream>>>(
        yA, ysB, st2, stS, PRM(k, 6), PRM(k, 7), PRM(k, 10), PRM(k, 11),
        (float*)d_out + e_off[k], sC, sHW, totalOut, invP);
  }
}
